// ResidualQuantizer_14456859918473
// MI455X (gfx1250) — matrix operand order repaired and checked by reading, not compiled, not run on hardware
//
#include <hip/hip_runtime.h>
#include <hip/hip_bf16.h>

typedef unsigned short u16;
typedef unsigned int u32;

#define B_SZ 32768
#define D_SZ 512
#define K_SZ 1024
#define L_SZ 4

typedef __bf16 bf16;
typedef bf16 v16bf __attribute__((ext_vector_type(16)));
typedef float v8f   __attribute__((ext_vector_type(8)));

union ABReg { uint4 q[2]; v16bf v; };

__device__ __forceinline__ u16 f2bf(float f) {
    u32 u = __float_as_uint(f);
    u += 0x7FFFu + ((u >> 16) & 1u);   // round-to-nearest-even
    return (u16)(u >> 16);
}

// ---------------------------------------------------------------------------
// Kernel 1: residual_f32 = z ; residual_bf16 = bf16(z) ; loss = 0
// ---------------------------------------------------------------------------
__global__ __launch_bounds__(256) void rq_init_kernel(
    const float* __restrict__ z,
    float* __restrict__ residF,
    u16*   __restrict__ residB16,
    float* __restrict__ lossPtr)
{
    const size_t i = ((size_t)blockIdx.x * 256 + threadIdx.x) * 4;
    float4 v = *reinterpret_cast<const float4*>(z + i);
    *reinterpret_cast<float4*>(residF + i) = v;
    ushort4 rb;
    rb.x = f2bf(v.x); rb.y = f2bf(v.y); rb.z = f2bf(v.z); rb.w = f2bf(v.w);
    *reinterpret_cast<ushort4*>(residB16 + i) = rb;
    if (blockIdx.x == 0 && threadIdx.x == 0) lossPtr[0] = 0.0f;
}

// ---------------------------------------------------------------------------
// Kernel 2: c2[row] = ||cb[row]||^2 ; cb_bf16 = bf16(cb).  One wave per row.
// ---------------------------------------------------------------------------
__global__ __launch_bounds__(256) void rq_c2_convert_kernel(
    const float* __restrict__ cb,     // [L*K, D]
    u16*   __restrict__ cbB16,        // [L*K, D]
    float* __restrict__ c2)           // [L*K]
{
    const int wave = threadIdx.x >> 5;
    const int lane = threadIdx.x & 31;
    const int row  = blockIdx.x * 8 + wave;       // 0 .. L*K-1
    const size_t base = (size_t)row * D_SZ;
    float sq = 0.0f;
    #pragma unroll
    for (int j = 0; j < D_SZ / 32; ++j) {
        float v = cb[base + j * 32 + lane];
        sq += v * v;
        cbB16[base + j * 32 + lane] = f2bf(v);
    }
    #pragma unroll
    for (int m = 1; m < 32; m <<= 1) sq += __shfl_xor(sq, m, 32);
    if (lane == 0) c2[row] = sq;
}

// ---------------------------------------------------------------------------
// Kernel 3 (per level): fused bf16-WMMA distance GEMM + row argmin.
// One block = 16 rows of B; 8 waves; each wave covers 8 column-tiles of 16.
// score[b,k] = ||c_k||^2 - 2 * <r_b, c_k>   (||r_b||^2 dropped: row-constant)
// ---------------------------------------------------------------------------
__global__ __launch_bounds__(256) void rq_argmin_kernel(
    const u16*   __restrict__ residB16,   // [B, D] bf16
    const u16*   __restrict__ cbB16,      // [K, D] bf16 (level slice; == col-major B)
    const float* __restrict__ c2,         // [K]
    int*         __restrict__ idxOut)     // [B]
{
    const int tid    = threadIdx.x;
    const int wave   = tid >> 5;
    const int lane   = tid & 31;
    const int lane15 = lane & 15;
    const int hi     = lane >> 4;
    const int row    = blockIdx.x * 16 + lane15;

    const u16* aRow = residB16 + (size_t)row * D_SZ;

    float bestV[8];
    int   bestN[8];
    #pragma unroll
    for (int v = 0; v < 8; ++v) { bestV[v] = 3.0e38f; bestN[v] = 0; }

    for (int j = 0; j < 8; ++j) {
        const int n0 = ((j << 3) + wave) << 4;    // base column of 16x16 tile
        const int n  = n0 + lane15;               // this lane's column
        const u16* bRow = cbB16 + (size_t)n * D_SZ;
        v8f acc = {0.f, 0.f, 0.f, 0.f, 0.f, 0.f, 0.f, 0.f};
        #pragma unroll
        for (int ks = 0; ks < D_SZ / 32; ++ks) {  // 16 WMMAs over K
            const int d0 = ks << 5;
            ABReg a, b;
            // 16-bit A 16x32 layout: lane half 'hi' holds K runs [hi*8, hi*8+8) and [16+hi*8, 24+hi*8)
            a.q[0] = *reinterpret_cast<const uint4*>(aRow + d0 + hi * 8);
            a.q[1] = *reinterpret_cast<const uint4*>(aRow + d0 + 16 + hi * 8);
            // 16-bit B 32x16 layout: lane half 'hi' holds K runs [hi*8, hi*8+8) and [16+hi*8, 24+hi*8)
            b.q[0] = *reinterpret_cast<const uint4*>(bRow + d0 + hi * 8);
            b.q[1] = *reinterpret_cast<const uint4*>(bRow + d0 + 16 + hi * 8);
            acc = __builtin_amdgcn_wmma_f32_16x16x32_bf16(
                false, a.v, false, b.v, (short)0, acc, false, false);
        }
        const float c2n = c2[n];
        #pragma unroll
        for (int v = 0; v < 8; ++v) {             // lane's rows: M = v + 8*hi
            float s = c2n - 2.0f * acc[v];
            if (s < bestV[v] || (s == bestV[v] && n < bestN[v])) {
                bestV[v] = s; bestN[v] = n;
            }
        }
    }

    // reduce across the 16 lanes of each half (same rows, different columns)
    #pragma unroll
    for (int v = 0; v < 8; ++v) {
        #pragma unroll
        for (int m = 1; m < 16; m <<= 1) {
            float ov = __shfl_xor(bestV[v], m, 32);
            int   on = __shfl_xor(bestN[v], m, 32);
            if (ov < bestV[v] || (ov == bestV[v] && on < bestN[v])) {
                bestV[v] = ov; bestN[v] = on;
            }
        }
    }

    __shared__ float sV[8 * 16];
    __shared__ int   sN[8 * 16];
    if (lane15 == 0) {
        #pragma unroll
        for (int v = 0; v < 8; ++v) {
            sV[wave * 16 + hi * 8 + v] = bestV[v];
            sN[wave * 16 + hi * 8 + v] = bestN[v];
        }
    }
    __syncthreads();
    if (tid < 16) {
        float bv = sV[tid]; int bn = sN[tid];
        #pragma unroll
        for (int w = 1; w < 8; ++w) {
            float ov = sV[w * 16 + tid]; int on = sN[w * 16 + tid];
            if (ov < bv || (ov == bv && on < bn)) { bv = ov; bn = on; }
        }
        idxOut[blockIdx.x * 16 + tid] = bn;
    }
}

// ---------------------------------------------------------------------------
// Kernel 4 (per level): gather chosen code, update residual (f32 + bf16),
// write level output, accumulate loss; last level also writes quantized.
// One block (128 threads * float4) per row.
// ---------------------------------------------------------------------------
__global__ __launch_bounds__(128) void rq_gather_update_kernel(
    const float* __restrict__ z,
    const float* __restrict__ cb,       // [K, D] level slice (f32)
    const int*   __restrict__ idx,      // [B]
    float* __restrict__ residF,
    u16*   __restrict__ residB16,
    float* __restrict__ lvlOut,         // d_out + (1+l)*B*D
    float* __restrict__ quantOut,       // d_out
    float* __restrict__ lossPtr,
    int isLast)
{
    const int b   = blockIdx.x;
    const int tid = threadIdx.x;
    const int d   = tid * 4;
    const int k   = idx[b];
    const size_t rOff = (size_t)b * D_SZ + d;

    float4 c = *reinterpret_cast<const float4*>(cb + (size_t)k * D_SZ + d);
    float4 r = *reinterpret_cast<const float4*>(residF + rOff);
    r.x -= c.x; r.y -= c.y; r.z -= c.z; r.w -= c.w;
    *reinterpret_cast<float4*>(residF + rOff) = r;

    ushort4 rb;
    rb.x = f2bf(r.x); rb.y = f2bf(r.y); rb.z = f2bf(r.z); rb.w = f2bf(r.w);
    *reinterpret_cast<ushort4*>(residB16 + rOff) = rb;

    *reinterpret_cast<float4*>(lvlOut + rOff) = c;

    if (isLast) {
        float4 zv = *reinterpret_cast<const float4*>(z + rOff);
        float4 q; q.x = zv.x - r.x; q.y = zv.y - r.y; q.z = zv.z - r.z; q.w = zv.w - r.w;
        *reinterpret_cast<float4*>(quantOut + rOff) = q;
    }

    float local = r.x * r.x + r.y * r.y + r.z * r.z + r.w * r.w;
    __shared__ float red[128];
    red[tid] = local;
    __syncthreads();
    for (int s = 64; s > 0; s >>= 1) {
        if (tid < s) red[tid] += red[tid + s];
        __syncthreads();
    }
    if (tid == 0)
        atomicAdd(lossPtr, red[0] * (1.0f / ((float)B_SZ * (float)D_SZ)));
}

// ---------------------------------------------------------------------------
extern "C" void kernel_launch(void* const* d_in, const int* in_sizes, int n_in,
                              void* d_out, int out_size, void* d_ws, size_t ws_size,
                              hipStream_t stream) {
    const float* z  = (const float*)d_in[0];   // [B, D]
    const float* cb = (const float*)d_in[1];   // [L, K, D]
    float* out = (float*)d_out;                // quantized | level_vecs | loss

    // workspace layout
    char* w = (char*)d_ws;
    const size_t residFBytes  = (size_t)B_SZ * D_SZ * 4;            // 64 MB
    const size_t residBBytes  = (size_t)B_SZ * D_SZ * 2;            // 32 MB
    const size_t cbBBytes     = (size_t)L_SZ * K_SZ * D_SZ * 2;     //  4 MB
    const size_t c2Bytes      = (size_t)L_SZ * K_SZ * 4;            // 16 KB
    float* residF   = (float*)w;
    u16*   residB16 = (u16*)(w + residFBytes);
    u16*   cbB16    = (u16*)(w + residFBytes + residBBytes);
    float* c2       = (float*)(w + residFBytes + residBBytes + cbBBytes);
    int*   idxWs    = (int*)(w + residFBytes + residBBytes + cbBBytes + c2Bytes);

    float* lossPtr = out + (size_t)(1 + L_SZ) * B_SZ * D_SZ;

    rq_init_kernel<<<(B_SZ * D_SZ) / 1024, 256, 0, stream>>>(z, residF, residB16, lossPtr);
    rq_c2_convert_kernel<<<(L_SZ * K_SZ) / 8, 256, 0, stream>>>(cb, cbB16, c2);

    for (int l = 0; l < L_SZ; ++l) {
        rq_argmin_kernel<<<B_SZ / 16, 256, 0, stream>>>(
            residB16, cbB16 + (size_t)l * K_SZ * D_SZ, c2 + l * K_SZ, idxWs);
        rq_gather_update_kernel<<<B_SZ, 128, 0, stream>>>(
            z, cb + (size_t)l * K_SZ * D_SZ, idxWs, residF, residB16,
            out + (size_t)(1 + l) * B_SZ * D_SZ, out, lossPtr, (l == L_SZ - 1) ? 1 : 0);
    }
}
